// HMM_43310450213038
// MI455X (gfx1250) — compile-verified, matrix operand order checked by reading, not run on hardware
//
#include <hip/hip_runtime.h>

// HMM forward (log-semiring chain reduction), CDNA5/gfx1250.
// Pass 1: per-(batch,chunk) workgroup folds L=64 64x64 log-factor matrices into
//         one chunk matrix via shifted-exp-domain matmuls on
//         v_wmma_f32_16x16x32_bf16 (f32 accumulate). F tiles stream HBM->LDS via
//         global_load_async_to_lds_b128 (double buffered); running matrix
//         renormalization is deferred one step so no cross-lane reductions.
// Pass 2: per-batch vector recursion over the 64 chunk matrices + final LSE.

#define HMM_B 8
#define HMM_T 4096
#define HMM_S 64
#define HMM_C 64              // chunks per batch (workspace: B*C*S*S*4 = 8 MB)
#define HMM_L (HMM_T / HMM_C) // 64 time steps per chunk

typedef __attribute__((ext_vector_type(16))) __bf16 v16bf;
typedef __attribute__((ext_vector_type(8)))  float  v8f;

union FragU { v16bf v; uint4 u[2]; };
union H8    { uint4 u; __bf16 h[8]; };

#if defined(__gfx1250__) && \
    __has_builtin(__builtin_amdgcn_global_load_async_to_lds_b128) && \
    __has_builtin(__builtin_amdgcn_s_wait_asynccnt)
#define HMM_ASYNC 1
// Builtin signature (from hipcc diagnostic): param0 = AS1 ptr to
// __attribute__((vector_size(16))) int, param1 = LDS ptr, then imm offset, cpol.
typedef int v4i_ls __attribute__((vector_size(16)));
typedef __attribute__((address_space(1))) v4i_ls GV4;
typedef __attribute__((address_space(3))) v4i_ls LV4;
#else
#define HMM_ASYNC 0
#endif

// A/B fragment per ISA 16-bit layout: lane holds 8 contiguous K-halfs at
// kbase and 8 more at kbase+16 -> two 16-byte LDS loads.
__device__ __forceinline__ v16bf load_frag(const __bf16* rowptr, int kbase) {
  FragU f;
  f.u[0] = *(const uint4*)(rowptr + kbase);
  f.u[1] = *(const uint4*)(rowptr + kbase + 16);
  return f.v;
}

// Per-thread 64B global->LDS tile copy (4 x b128).
__device__ __forceinline__ void issue_tile(const float* src, float* dst) {
#if HMM_ASYNC
  GV4* g = (GV4*)src;
  LV4* l = (LV4*)dst;
  __builtin_amdgcn_global_load_async_to_lds_b128(g, l, 0, 0);
  __builtin_amdgcn_global_load_async_to_lds_b128(g, l, 16, 0);
  __builtin_amdgcn_global_load_async_to_lds_b128(g, l, 32, 0);
  __builtin_amdgcn_global_load_async_to_lds_b128(g, l, 48, 0);
#else
  const float4* s = (const float4*)src;
  float4* d = (float4*)dst;
  d[0] = s[0]; d[1] = s[1]; d[2] = s[2]; d[3] = s[3];
#endif
}

__launch_bounds__(256)
__global__ void hmm_chunk_kernel(const float* __restrict__ logF,
                                 float* __restrict__ chunkM) {
  __shared__ __align__(16) float  sF[2][HMM_S][68]; // double-buffered F_t (f32)
  __shared__ __align__(16) __bf16 sA[HMM_S][72];    // running matrix, exp dom, [i][k]
  __shared__ __align__(16) __bf16 sB[HMM_S][72];    // exp(F - colmax), transposed [j][k]
  __shared__ float sQ [HMM_S][4];                   // colmax(F) partials
  __shared__ float sQ2[HMM_S][4];                   // rowmax(A) partials
  __shared__ float sr[HMM_S];                       // per-row log offsets
  __shared__ float sc[HMM_S];                       // colmax of F_t
  __shared__ float sec[HMM_S];                      // exp(colmax)
  __shared__ float sInvU[HMM_S];                    // 1 / rowmax(A)

  const int tid  = threadIdx.x;
  const int b    = blockIdx.y;
  const int cchk = blockIdx.x;
  const float* Fp = logF + (((size_t)b * HMM_T) + (size_t)cchk * HMM_L) * (HMM_S * HMM_S);

  const int row = tid >> 2;   // 0..63
  const int seg = tid & 3;    // 0..3 (16-element segment)
  const int wave = tid >> 5;
  const int lane = tid & 31;
  const int l16  = lane & 15;
  const int hi8  = (lane >> 4) << 3;  // 0 or 8

  const float* tsrc = Fp + row * HMM_S + seg * 16;  // per-thread 64B segment of F_t
  float* dst0 = &sF[0][row][seg * 16];
  float* dst1 = &sF[1][row][seg * 16];

  // ---- stream F_0, start F_1 ----
  issue_tile(tsrc, dst0);
#if HMM_ASYNC
  __builtin_amdgcn_s_wait_asynccnt(0);
#endif
  __syncthreads();
  issue_tile(tsrc + (HMM_S * HMM_S), dst1);
#if !HMM_ASYNC
  __syncthreads();
#endif

  // ---- init: running matrix = F_0, row-shifted into exp domain ----
  {
    float m = -INFINITY;
    #pragma unroll
    for (int j = 0; j < 16; ++j) m = fmaxf(m, sF[0][row][seg * 16 + j]);
    sQ[row][seg] = m;
  }
  __syncthreads();
  if (tid < HMM_S)
    sr[tid] = fmaxf(fmaxf(sQ[tid][0], sQ[tid][1]), fmaxf(sQ[tid][2], sQ[tid][3]));
  __syncthreads();
  {
    const float r = sr[row];
    H8 p0, p1;
    #pragma unroll
    for (int kk = 0; kk < 8; ++kk)
      p0.h[kk] = (__bf16)__expf(sF[0][row][seg * 16 + kk] - r);
    #pragma unroll
    for (int kk = 0; kk < 8; ++kk)
      p1.h[kk] = (__bf16)__expf(sF[0][row][seg * 16 + 8 + kk] - r);
    *(uint4*)&sA[row][seg * 16]     = p0.u;
    *(uint4*)&sA[row][seg * 16 + 8] = p1.u;
  }

  // ---- fold remaining L-1 matrices:  M <- M (x) F_t ----
  for (int t = 1; t < HMM_L; ++t) {
    const int cb = t & 1;
#if HMM_ASYNC
    __builtin_amdgcn_s_wait_asynccnt(0);
#endif
    __syncthreads();  // F_t resident; prev-step sA writes visible
    if (t + 1 < HMM_L)
      issue_tile(tsrc + (size_t)(t + 1) * (HMM_S * HMM_S), cb ? dst0 : dst1);
#if !HMM_ASYNC
    if (t + 1 < HMM_L)
      __builtin_prefetch(tsrc + (size_t)(t + 1) * (HMM_S * HMM_S), 0, 1);
    __syncthreads();
#endif

    // staging: colmax(F_t) partials + rowmax(A) partials, fully lane-parallel
    {
      const int j = tid & 63, part = tid >> 6;
      float m = -INFINITY;
      #pragma unroll
      for (int kk = 0; kk < 16; ++kk) m = fmaxf(m, sF[cb][part * 16 + kk][j]);
      sQ[j][part] = m;

      H8 a0, a1;
      a0.u = *(const uint4*)&sA[j][part * 16];
      a1.u = *(const uint4*)&sA[j][part * 16 + 8];
      float um = -INFINITY;
      #pragma unroll
      for (int x = 0; x < 8; ++x)
        um = fmaxf(um, fmaxf((float)a0.h[x], (float)a1.h[x]));
      sQ2[j][part] = um;
    }
    __syncthreads();
    if (tid < HMM_S) {
      const float cj = fmaxf(fmaxf(sQ[tid][0], sQ[tid][1]), fmaxf(sQ[tid][2], sQ[tid][3]));
      sc[tid]  = cj;
      sec[tid] = __expf(cj);
    } else if (tid < 2 * HMM_S) {
      const int i = tid - HMM_S;
      const float u = fmaxf(fmaxf(sQ2[i][0], sQ2[i][1]), fmaxf(sQ2[i][2], sQ2[i][3]));
      sInvU[i] = 1.0f / u;
      sr[i] += __logf(u);   // exact: scale of A folds into log offset
    }
    __syncthreads();

    // B = exp(F - colmax), transposed [j][k], packed b128 stores
    {
      const float cj = sc[row];
      H8 p0, p1;
      #pragma unroll
      for (int kk = 0; kk < 8; ++kk)
        p0.h[kk] = (__bf16)__expf(sF[cb][seg * 16 + kk][row] - cj);
      #pragma unroll
      for (int kk = 0; kk < 8; ++kk)
        p1.h[kk] = (__bf16)__expf(sF[cb][seg * 16 + 8 + kk][row] - cj);
      *(uint4*)&sB[row][seg * 16]     = p0.u;
      *(uint4*)&sB[row][seg * 16 + 8] = p1.u;
    }
    __syncthreads();

    // P^T = B^T @ A^T : WMMA-A operand = sB rows (j,k), WMMA-B operand = sA (k,i).
    // D[j,i] -> lane owns column i, 8 contiguous j per tile => packed A' store.
    v8f acc[2];
    int itc[2], jtc[2];
    #pragma unroll
    for (int q = 0; q < 2; ++q) {
      const int tt = wave * 2 + q;
      jtc[q] = tt >> 2; itc[q] = tt & 3;
      v8f a = {};
      #pragma unroll
      for (int kb = 0; kb < 2; ++kb) {
        const int kbase = kb * 32 + hi8;
        const v16bf bfrag = load_frag(&sB[jtc[q] * 16 + l16][0], kbase);
        const v16bf afrag = load_frag(&sA[itc[q] * 16 + l16][0], kbase);
        a = __builtin_amdgcn_wmma_f32_16x16x32_bf16(false, bfrag, false, afrag,
                                                    (short)0, a, false, false);
      }
      acc[q] = a;
    }
    __syncthreads();  // all fragment reads done before sA overwrite

    // A'[i][j] = (P[i][j]/u_i) * exp(c_j); bounded ~64*e^cmax, bf16-safe
    #pragma unroll
    for (int q = 0; q < 2; ++q) {
      const int i  = itc[q] * 16 + l16;
      const int jb = jtc[q] * 16 + hi8;
      const float invu = sInvU[i];
      H8 pk;
      #pragma unroll
      for (int g = 0; g < 8; ++g)
        pk.h[g] = (__bf16)(acc[q][g] * invu * sec[jb + g]);
      *(uint4*)&sA[i][jb] = pk.u;
    }
  }
  __syncthreads();

  // ---- emit chunk matrix back in log domain ----
  {
    const float r = sr[row];
    float* dst = chunkM + (((size_t)b * HMM_C + cchk) << 12) + row * HMM_S + seg * 16;
    #pragma unroll
    for (int j = 0; j < 16; ++j)
      dst[j] = r + __logf((float)sA[row][seg * 16 + j]);
  }
}

__launch_bounds__(64)
__global__ void hmm_finalize_kernel(const float* __restrict__ chunkM,
                                    const float* __restrict__ logInit,
                                    float* __restrict__ out) {
  __shared__ float sa[HMM_S];
  const int j = threadIdx.x;
  const int b = blockIdx.x;
  sa[j] = logInit[b * HMM_S + j];
  __syncthreads();
  for (int c = 0; c < HMM_C; ++c) {
    const float* M = chunkM + (((size_t)b * HMM_C + c) << 12);
    float m = -INFINITY;
    for (int i = 0; i < HMM_S; ++i) m = fmaxf(m, sa[i] + M[i * HMM_S + j]);
    float s = 0.f;
    for (int i = 0; i < HMM_S; ++i) s += __expf(sa[i] + M[i * HMM_S + j] - m);
    const float na = m + __logf(s);
    __syncthreads();
    sa[j] = na;
    __syncthreads();
  }
  float m = -INFINITY;
  for (int i = 0; i < HMM_S; ++i) m = fmaxf(m, sa[i]);
  float s = 0.f;
  for (int i = 0; i < HMM_S; ++i) s += __expf(sa[i] - m);
  if (j == 0) out[b] = m + __logf(s);
}

extern "C" void kernel_launch(void* const* d_in, const int* in_sizes, int n_in,
                              void* d_out, int out_size, void* d_ws, size_t ws_size,
                              hipStream_t stream) {
  (void)in_sizes; (void)n_in; (void)out_size; (void)ws_size;
  const float* logF    = (const float*)d_in[0];
  const float* logInit = (const float*)d_in[1];
  float* out    = (float*)d_out;
  float* chunkM = (float*)d_ws;  // B*C*64*64 f32 = 8 MB scratch

  dim3 grid1(HMM_C, HMM_B);
  hmm_chunk_kernel<<<grid1, 256, 0, stream>>>(logF, chunkM);
  hmm_finalize_kernel<<<HMM_B, HMM_S, 0, stream>>>(chunkM, logInit, out);
}
